// AdditiveAttention_71777493451210
// MI455X (gfx1250) — matrix operand order repaired and checked by reading, not compiled, not run on hardware
//
#include <hip/hip_runtime.h>
#include <math.h>

#define B_ 16
#define S_ 2048
#define D_ 1024

typedef __attribute__((ext_vector_type(16))) __bf16 v16bf;
typedef __attribute__((ext_vector_type(2)))  __bf16 v2bf;
typedef __attribute__((ext_vector_type(8)))  float  v8f;

union Frag {
  uint4 u[2];
  v16bf h;
};

__device__ __forceinline__ unsigned short f32_to_bf16(float f) {
  unsigned int u = __float_as_uint(f);
  u += 0x7FFFu + ((u >> 16) & 1u);   // round-to-nearest-even
  return (unsigned short)(u >> 16);
}

// Packed f32x2 -> bf16x2 (hardware v_cvt_pk_bf16_f32 when available)
__device__ __forceinline__ unsigned int pack_bf16(float a, float b) {
#if __has_builtin(__builtin_amdgcn_cvt_pk_bf16_f32)
  union { v2bf h; unsigned int u; } cv;
  cv.h = __builtin_amdgcn_cvt_pk_bf16_f32(a, b);
  return cv.u;
#else
  return (unsigned int)f32_to_bf16(a) | ((unsigned int)f32_to_bf16(b) << 16);
#endif
}

// Hardware V_TANH_F32 when the builtin exists; OCML fallback otherwise.
__device__ __forceinline__ float fast_tanh(float x) {
#if __has_builtin(__builtin_amdgcn_tanhf)
  return __builtin_amdgcn_tanhf(x);
#elif __has_builtin(__builtin_amdgcn_tanh_f32)
  return __builtin_amdgcn_tanh_f32(x);
#else
  return tanhf(x);
#endif
}

// ---------------------------------------------------------------------------
// Kernel 1: Wk [d][e] f32  ->  WkT [e][d] bf16   (N-major for WMMA B frags)
// ---------------------------------------------------------------------------
__global__ __launch_bounds__(256) void k_wkT(const float* __restrict__ Wk,
                                             unsigned short* __restrict__ WkT) {
  __shared__ float tile[32][33];
  const int bx = blockIdx.x * 32;  // e
  const int by = blockIdx.y * 32;  // d
  const int tx = threadIdx.x, ty = threadIdx.y;
  #pragma unroll
  for (int i = ty; i < 32; i += 8)
    tile[i][tx] = Wk[(size_t)(by + i) * D_ + bx + tx];
  __syncthreads();
  // Each thread writes 2 consecutive d (packed 32-bit store, coalesced).
  unsigned int* outrow = reinterpret_cast<unsigned int*>(WkT);
  #pragma unroll
  for (int i = ty; i < 32; i += 8) {
    // e = bx + i + (tx>=16? handled by second half): use tx&15 as d-pair
    // d pair = by + 2*(tx&15), e = bx + i + 16*(tx>>4)
    unsigned int p = pack_bf16(tile[2 * (tx & 15)][i + 16 * (tx >> 4)],
                               tile[2 * (tx & 15) + 1][i + 16 * (tx >> 4)]);
    outrow[((size_t)(bx + i + 16 * (tx >> 4)) * D_ + by) / 2 + (tx & 15)] = p;
  }
}

// ---------------------------------------------------------------------------
// Kernel 2: cvec[b][e] = sum_d q[b][d] * Wq[d][e] + bias[e]
// ---------------------------------------------------------------------------
__global__ __launch_bounds__(256) void k_qproj(const float* __restrict__ q,
                                               const float* __restrict__ Wq,
                                               const float* __restrict__ bias,
                                               float* __restrict__ cvec) {
  const int b = blockIdx.y;
  const int e = blockIdx.x * 256 + threadIdx.x;
  float acc = 0.f;
  for (int d = 0; d < D_; ++d)
    acc = fmaf(q[b * D_ + d], Wq[(size_t)d * D_ + e], acc);
  cvec[b * D_ + e] = acc + bias[e];
}

// ---------------------------------------------------------------------------
// Kernel 3: fused score = w_score . tanh(k@Wk + cvec) + b_score
// One WG = 32 rows of k (2 WMMA M-subtiles). 8 waves, each owns 128 N cols.
// 2 N-tiles blocked per pass: 4 WMMAs share 2 A frags + 2 B frags.
// ---------------------------------------------------------------------------
__global__ __launch_bounds__(256) void k_score(const float* __restrict__ kmat,
                                               const unsigned short* __restrict__ WkT,
                                               const float* __restrict__ cvec,
                                               const float* __restrict__ w_score,
                                               const float* __restrict__ b_score,
                                               float* __restrict__ score_out) {
  __shared__ unsigned short kbuf[32][1032];  // bf16, padded stride (16B aligned rows)
  __shared__ float ssc[32][8];               // [row][wave]

  const int b = blockIdx.y;
  const int srow0 = blockIdx.x * 32;
  const int tid = threadIdx.x;

  // Stage 32 x 1024 f32 rows of k as bf16 into LDS (float4 loads, 8B stores).
  const float4* kb4 =
      reinterpret_cast<const float4*>(kmat + ((size_t)b * S_ + srow0) * D_);
  #pragma unroll 8
  for (int it = 0; it < 32; ++it) {
    int i = tid + it * 256;
    int row = i >> 8;      // 256 float4 per row
    int c4 = i & 255;
    float4 f = kb4[row * 256 + c4];
    *reinterpret_cast<uint2*>(&kbuf[row][c4 * 4]) =
        make_uint2(pack_bf16(f.x, f.y), pack_bf16(f.z, f.w));
  }
  __syncthreads();

  const int wave = tid >> 5;
  const int lane = tid & 31;
  const int half = lane >> 4;   // lane half selects K sub-range per ISA layout
  const int ln   = lane & 15;

  float part0[8], part1[8];
  #pragma unroll
  for (int r = 0; r < 8; ++r) { part0[r] = 0.f; part1[r] = 0.f; }

  // A frag (16x32 bf16): u[0] @ K = kc + 16*half, u[1] @ K = kc + 16*half + 8
  const unsigned short* a0base = &kbuf[ln][16 * half];
  const unsigned short* a1base = &kbuf[16 + ln][16 * half];

  #pragma unroll 1
  for (int nt = 0; nt < 8; nt += 2) {
    const int n0 = wave * 128 + nt * 16 + ln;
    const int n1 = n0 + 16;
    const float c0 = cvec[b * D_ + n0];
    const float c1 = cvec[b * D_ + n1];
    const float w0 = w_score[n0];
    const float w1 = w_score[n1];
    // B frag (32x16 bf16): 32 contiguous bytes of WkT row n at K = kc + 16*half
    const unsigned short* b0base = WkT + (size_t)n0 * D_ + 16 * half;
    const unsigned short* b1base = WkT + (size_t)n1 * D_ + 16 * half;

    v8f acc00 = {}, acc01 = {}, acc10 = {}, acc11 = {};
    #pragma unroll 2
    for (int kc = 0; kc < D_; kc += 32) {
      Frag fb0, fb1, fa0, fa1;
      fb0.u[0] = ((const uint4*)(b0base + kc))[0];
      fb0.u[1] = ((const uint4*)(b0base + kc))[1];
      fb1.u[0] = ((const uint4*)(b1base + kc))[0];
      fb1.u[1] = ((const uint4*)(b1base + kc))[1];
      fa0.u[0] = *(const uint4*)(a0base + kc);
      fa0.u[1] = *(const uint4*)(a0base + kc + 8);
      fa1.u[0] = *(const uint4*)(a1base + kc);
      fa1.u[1] = *(const uint4*)(a1base + kc + 8);
      acc00 = __builtin_amdgcn_wmma_f32_16x16x32_bf16(false, fa0.h, false, fb0.h,
                                                      (short)0, acc00, false, false);
      acc10 = __builtin_amdgcn_wmma_f32_16x16x32_bf16(false, fa1.h, false, fb0.h,
                                                      (short)0, acc10, false, false);
      acc01 = __builtin_amdgcn_wmma_f32_16x16x32_bf16(false, fa0.h, false, fb1.h,
                                                      (short)0, acc01, false, false);
      acc11 = __builtin_amdgcn_wmma_f32_16x16x32_bf16(false, fa1.h, false, fb1.h,
                                                      (short)0, acc11, false, false);
    }
    // Fused epilogue: hardware tanh + dot with w_score (C row = r + 8*half)
    #pragma unroll
    for (int r = 0; r < 8; ++r) {
      part0[r] += fast_tanh(acc00[r] + c0) * w0 + fast_tanh(acc01[r] + c1) * w1;
      part1[r] += fast_tanh(acc10[r] + c0) * w0 + fast_tanh(acc11[r] + c1) * w1;
    }
  }

  // Deterministic reduction: xor-shuffle within each 16-lane half, then across waves.
  #pragma unroll
  for (int r = 0; r < 8; ++r) {
    float v0 = part0[r], v1 = part1[r];
    #pragma unroll
    for (int m = 1; m < 16; m <<= 1) {
      v0 += __shfl_xor(v0, m);
      v1 += __shfl_xor(v1, m);
    }
    if (ln == 0) {
      ssc[8 * half + r][wave] = v0;        // rows 0..15 (subtile 0)
      ssc[16 + 8 * half + r][wave] = v1;   // rows 16..31 (subtile 1)
    }
  }
  __syncthreads();
  if (tid < 32) {
    float s = 0.f;
    #pragma unroll
    for (int w = 0; w < 8; ++w) s += ssc[tid][w];
    score_out[(size_t)b * S_ + srow0 + tid] = s + b_score[0];
  }
}

// ---------------------------------------------------------------------------
// Kernel 4: in-place softmax over S per batch (scores live in d_out attn region)
// ---------------------------------------------------------------------------
__global__ __launch_bounds__(256) void k_softmax(float* __restrict__ attn) {
  const int b = blockIdx.x;
  const int tid = threadIdx.x;
  float* row = attn + (size_t)b * S_;
  __shared__ float red[8];

  float vals[8];
  float lmax = -3.402823466e38f;
  #pragma unroll
  for (int i = 0; i < 8; ++i) {
    vals[i] = row[tid + i * 256];
    lmax = fmaxf(lmax, vals[i]);
  }
  #pragma unroll
  for (int m = 16; m >= 1; m >>= 1) lmax = fmaxf(lmax, __shfl_xor(lmax, m));
  if ((tid & 31) == 0) red[tid >> 5] = lmax;
  __syncthreads();
  float bmax = red[0];
  #pragma unroll
  for (int w = 1; w < 8; ++w) bmax = fmaxf(bmax, red[w]);

  float lsum = 0.f;
  #pragma unroll
  for (int i = 0; i < 8; ++i) {
    vals[i] = __expf(vals[i] - bmax);
    lsum += vals[i];
  }
  #pragma unroll
  for (int m = 16; m >= 1; m >>= 1) lsum += __shfl_xor(lsum, m);
  __syncthreads();
  if ((tid & 31) == 0) red[tid >> 5] = lsum;
  __syncthreads();
  float bsum = 0.f;
  #pragma unroll
  for (int w = 0; w < 8; ++w) bsum += red[w];
  const float inv = 1.f / bsum;
  #pragma unroll
  for (int i = 0; i < 8; ++i) row[tid + i * 256] = vals[i] * inv;
}

// ---------------------------------------------------------------------------
// Kernel 5: context partials: part[b][sc][:] = sum_{s in chunk} attn[b,s]*v[b,s,:]
// 128 WGs for HBM parallelism; fixed-order accumulation (deterministic).
// ---------------------------------------------------------------------------
__global__ __launch_bounds__(256) void k_ctx_part(const float* __restrict__ v,
                                                  const float* __restrict__ attn,
                                                  float* __restrict__ part) {
  const int b = blockIdx.y, sc = blockIdx.x, tid = threadIdx.x;
  const float4* vb =
      reinterpret_cast<const float4*>(v + ((size_t)b * S_ + sc * 256) * D_);
  const float* ab = attn + (size_t)b * S_ + sc * 256;
  float4 acc = make_float4(0.f, 0.f, 0.f, 0.f);
  for (int s = 0; s < 256; ++s) {
    const float a = ab[s];
    const float4 x = vb[(size_t)s * 256 + tid];
    acc.x = fmaf(a, x.x, acc.x);
    acc.y = fmaf(a, x.y, acc.y);
    acc.z = fmaf(a, x.z, acc.z);
    acc.w = fmaf(a, x.w, acc.w);
  }
  reinterpret_cast<float4*>(part + ((size_t)(b * 8 + sc)) * D_)[tid] = acc;
}

__global__ __launch_bounds__(256) void k_ctx_reduce(const float* __restrict__ part,
                                                    float* __restrict__ ctx) {
  const int b = blockIdx.x, tid = threadIdx.x;
  float4 acc = make_float4(0.f, 0.f, 0.f, 0.f);
  #pragma unroll
  for (int sc = 0; sc < 8; ++sc) {
    const float4 x =
        reinterpret_cast<const float4*>(part + ((size_t)(b * 8 + sc)) * D_)[tid];
    acc.x += x.x; acc.y += x.y; acc.z += x.z; acc.w += x.w;
  }
  reinterpret_cast<float4*>(ctx + (size_t)b * D_)[tid] = acc;
}

// ---------------------------------------------------------------------------
extern "C" void kernel_launch(void* const* d_in, const int* in_sizes, int n_in,
                              void* d_out, int out_size, void* d_ws, size_t ws_size,
                              hipStream_t stream) {
  (void)in_sizes; (void)n_in; (void)out_size; (void)ws_size;
  const float* q       = (const float*)d_in[0];
  const float* k       = (const float*)d_in[1];
  const float* v       = (const float*)d_in[2];
  const float* Wq      = (const float*)d_in[3];
  const float* Wk      = (const float*)d_in[4];
  const float* bias    = (const float*)d_in[5];
  const float* w_score = (const float*)d_in[6];
  const float* b_score = (const float*)d_in[7];

  float* ctx  = (float*)d_out;            // [16,1,1024]
  float* attn = (float*)d_out + B_ * D_;  // [16,2048] (holds scores, then attn)

  // Workspace layout: WkT bf16 (2 MB) | cvec f32 (64 KB) | partials f32 (512 KB)
  unsigned short* WkT = (unsigned short*)d_ws;
  float* cvec = (float*)((char*)d_ws + (size_t)D_ * D_ * 2);
  float* part = cvec + B_ * D_;

  k_wkT      <<<dim3(32, 32),     dim3(32, 8), 0, stream>>>(Wk, WkT);
  k_qproj    <<<dim3(4, B_),      256,         0, stream>>>(q, Wq, bias, cvec);
  k_score    <<<dim3(S_/32, B_),  256,         0, stream>>>(k, WkT, cvec, w_score, b_score, attn);
  k_softmax  <<<B_,               256,         0, stream>>>(attn);
  k_ctx_part <<<dim3(8, B_),      256,         0, stream>>>(v, attn, part);
  k_ctx_reduce<<<B_,              256,         0, stream>>>(part, ctx);
}
